// AttentionMHA_53996328845410
// MI455X (gfx1250) — matrix operand order repaired and checked by reading, not compiled, not run on hardware
//
#include <hip/hip_runtime.h>

// ---------------- problem constants ----------------
constexpr int S_   = 2048;
constexpr int D_   = 4096;
constexpr int H_   = 32;
constexpr int KVH_ = 8;
constexpr int HD_  = 128;
constexpr int NQKV_ = D_ + 2 * KVH_ * HD_;   // 6144
constexpr int CTX_  = 4096;

typedef __bf16 bf16_t;
typedef __attribute__((ext_vector_type(16))) __bf16 v16bf;
typedef __attribute__((ext_vector_type(8)))  float  v8f;

union FragU { v16bf v; uint4 u[2]; };
union Pack4 { bf16_t h[4]; uint2 u; };

__device__ inline v8f vzero8() {
  v8f z = {0.f, 0.f, 0.f, 0.f, 0.f, 0.f, 0.f, 0.f};
  return z;
}

__device__ inline v8f wmma_bf16(v16bf a, v16bf b, v8f c) {
  // D = A(16x32 bf16) * B(32x16 bf16) + C(16x16 f32)
  return __builtin_amdgcn_wmma_f32_16x16x32_bf16(
      false, a, false, b, (short)0, c, false, false);
}

// ---- CDNA5 async global->LDS copy (ASYNCcnt-tracked DMA), 16B per lane ----
__device__ inline void async_copy16(const bf16_t* __restrict__ g, bf16_t* l) {
  unsigned lds_addr = (unsigned)(unsigned long long)l;  // low 32 bits = LDS addr
  asm volatile("global_load_async_to_lds_b128 %0, %1, off"
               :: "v"(lds_addr), "v"(g) : "memory");
}
__device__ inline void async_wait_all() {
  asm volatile("s_wait_asynccnt 0x0" ::: "memory");
}

// ---- WMMA fragment loads (work for global or LDS pointers) ----------------
// A-fragment 16x32: lane<16: row=m0+lane, K chunks [k0..k0+7],[k0+16..k0+23]
//                   lane>=16: row=m0+lane-16, chunks [k0+8..15],[k0+24..31]
__device__ inline v16bf fragA(const bf16_t* base, int lda, int m0, int k0) {
  int lane = threadIdx.x & 31;
  const bf16_t* p = base + (size_t)(m0 + (lane & 15)) * lda + k0 + ((lane >> 4) << 3);
  FragU f;
  f.u[0] = *(const uint4*)p;
  f.u[1] = *(const uint4*)(p + 16);
  return f.v;
}
// B-fragment 32x16: col=n0+(lane&15); K rows [k0+8h..+7],[k0+16+8h..+7], h=lane>>4.
__device__ inline v16bf fragB(const bf16_t* base, int ldb, int k0, int n0) {
  int lane = threadIdx.x & 31;
  const bf16_t* p = base + (size_t)(k0 + ((lane >> 4) << 3)) * ldb + n0 + (lane & 15);
  v16bf v;
#pragma unroll
  for (int s = 0; s < 8; ++s) {
    v[s]     = p[(size_t)s * ldb];
    v[s + 8] = p[(size_t)(s + 16) * ldb];
  }
  return v;
}

// ---------------- fp32 -> bf16 cast/copy (optional column offset) ----------
__global__ void cvt_bf16_kernel(const float* __restrict__ src,
                                bf16_t* __restrict__ dst,
                                int total, int ncols, int ldd, int colOff) {
  int idx = blockIdx.x * blockDim.x + threadIdx.x;
  if (idx >= total) return;
  int r = idx / ncols;
  int c = idx - r * ncols;
  dst[(size_t)r * ldd + colOff + c] = (bf16_t)src[idx];
}

// ---------------- bf16 WMMA GEMM with async-LDS double buffering -----------
// block 256 (8 waves); block tile 128x128; BK=64; wave tile 32x64.
constexpr int BK    = 64;
constexpr int LDA_S = BK + 8;    // 72  (bank-spread pad)
constexpr int LDB_S = 128 + 8;   // 136

__global__ __launch_bounds__(256)
void gemm_bf16_kernel(const bf16_t* __restrict__ A, const bf16_t* __restrict__ B,
                      float* __restrict__ D, int M, int N, int K) {
  __shared__ __attribute__((aligned(16))) bf16_t Ab[2][128 * LDA_S];
  __shared__ __attribute__((aligned(16))) bf16_t Bb[2][BK * LDB_S];
  int wave = threadIdx.x >> 5;
  int lane = threadIdx.x & 31;
  int wm = wave & 3;             // 4 waves x 32 rows
  int wn = wave >> 2;            // 2 waves x 64 cols
  int mBlk = blockIdx.y * 128;
  int nBlk = blockIdx.x * 128;

  auto copy_panels = [&](int buf, int k0) {
    int t = threadIdx.x;
#pragma unroll
    for (int i = 0; i < 4; ++i) {
      int c = t + i * 256;                   // 0..1023
      int ra = c >> 3, sa = (c & 7) << 3;    // A: 128 rows x 64 elems
      async_copy16(A + (size_t)(mBlk + ra) * K + k0 + sa, &Ab[buf][ra * LDA_S + sa]);
      int rb = c >> 4, sb = (c & 15) << 3;   // B: 64 rows x 128 elems
      async_copy16(B + (size_t)(k0 + rb) * N + nBlk + sb, &Bb[buf][rb * LDB_S + sb]);
    }
  };

  v8f acc[2][4];
#pragma unroll
  for (int i = 0; i < 2; ++i)
#pragma unroll
    for (int j = 0; j < 4; ++j) acc[i][j] = vzero8();

  copy_panels(0, 0);
  async_wait_all();
  __syncthreads();

  int buf = 0;
  for (int k0 = 0; k0 < K; k0 += BK) {
    if (k0 + BK < K) copy_panels(buf ^ 1, k0 + BK);
#pragma unroll
    for (int ks = 0; ks < BK; ks += 32) {
      v16bf a0 = fragA(&Ab[buf][0], LDA_S, wm * 32, ks);
      v16bf a1 = fragA(&Ab[buf][0], LDA_S, wm * 32 + 16, ks);
      v16bf b0 = fragB(&Bb[buf][0], LDB_S, ks, wn * 64);
      v16bf b1 = fragB(&Bb[buf][0], LDB_S, ks, wn * 64 + 16);
      v16bf b2 = fragB(&Bb[buf][0], LDB_S, ks, wn * 64 + 32);
      v16bf b3 = fragB(&Bb[buf][0], LDB_S, ks, wn * 64 + 48);
      acc[0][0] = wmma_bf16(a0, b0, acc[0][0]);
      acc[0][1] = wmma_bf16(a0, b1, acc[0][1]);
      acc[0][2] = wmma_bf16(a0, b2, acc[0][2]);
      acc[0][3] = wmma_bf16(a0, b3, acc[0][3]);
      acc[1][0] = wmma_bf16(a1, b0, acc[1][0]);
      acc[1][1] = wmma_bf16(a1, b1, acc[1][1]);
      acc[1][2] = wmma_bf16(a1, b2, acc[1][2]);
      acc[1][3] = wmma_bf16(a1, b3, acc[1][3]);
    }
    async_wait_all();
    __syncthreads();
    buf ^= 1;
  }

  int rowadd = (lane >> 4) << 3;
  int coll   = lane & 15;
#pragma unroll
  for (int i = 0; i < 2; ++i)
#pragma unroll
    for (int j = 0; j < 4; ++j)
#pragma unroll
      for (int r = 0; r < 8; ++r) {
        int row = mBlk + wm * 32 + i * 16 + rowadd + r;
        int col = nBlk + wn * 64 + j * 16 + coll;
        D[(size_t)row * N + col] = acc[i][j][r];
      }
}

// ---------------- RoPE + RMSNorm + cache layout builders --------------------
// grid (6, S_), block 256.  wave -> slot 0..47: 0-31 q, 32-39 k, 40-47 v
__global__ __launch_bounds__(256)
void rope_norm_kernel(const float* __restrict__ qkv,
                      const float* __restrict__ fcos, const float* __restrict__ fsin,
                      const float* __restrict__ qw,   const float* __restrict__ kw,
                      bf16_t* __restrict__ qn,   // [H][S][HD]
                      bf16_t* __restrict__ kt,   // [KVH][HD][CTX] (transposed)
                      bf16_t* __restrict__ vn) { // [KVH][CTX][HD]
  int wave = threadIdx.x >> 5;
  int lane = threadIdx.x & 31;
  int head = blockIdx.x * 8 + wave;
  int s    = blockIdx.y;
  const float* row = qkv + (size_t)s * NQKV_;
  int d0 = lane * 4;

  if (head < H_ + KVH_) {
    bool isq = head < H_;
    int col = isq ? head * HD_ : D_ + (head - H_) * HD_;
    float4 tv = *(const float4*)(row + col + d0);
    float c0 = fcos[s * 64 + 2 * lane],     s0 = fsin[s * 64 + 2 * lane];
    float c1 = fcos[s * 64 + 2 * lane + 1], s1 = fsin[s * 64 + 2 * lane + 1];
    float o0 = tv.x * c0 - tv.y * s0;
    float o1 = tv.x * s0 + tv.y * c0;
    float o2 = tv.z * c1 - tv.w * s1;
    float o3 = tv.z * s1 + tv.w * c1;
    float ss = o0 * o0 + o1 * o1 + o2 * o2 + o3 * o3;
#pragma unroll
    for (int d = 1; d < 32; d <<= 1) ss += __shfl_xor(ss, d, 32);
    float rn = rsqrtf(ss * (1.0f / HD_) + 1e-5f);
    const float* w = isq ? qw : kw;
    float r0 = o0 * rn * w[d0 + 0];
    float r1 = o1 * rn * w[d0 + 1];
    float r2 = o2 * rn * w[d0 + 2];
    float r3 = o3 * rn * w[d0 + 3];
    if (isq) {
      Pack4 pk;
      pk.h[0] = (bf16_t)r0; pk.h[1] = (bf16_t)r1;
      pk.h[2] = (bf16_t)r2; pk.h[3] = (bf16_t)r3;
      *(uint2*)(qn + ((size_t)(head * S_ + s)) * HD_ + d0) = pk.u;
    } else {
      int kvh = head - H_;
      bf16_t* base = kt + (size_t)(kvh * HD_ + d0) * CTX_ + s;
      base[0 * CTX_] = (bf16_t)r0;
      base[1 * CTX_] = (bf16_t)r1;
      base[2 * CTX_] = (bf16_t)r2;
      base[3 * CTX_] = (bf16_t)r3;
    }
  } else {
    int kvh = head - H_ - KVH_;
    float4 tv = *(const float4*)(row + D_ + KVH_ * HD_ + kvh * HD_ + d0);
    Pack4 pk;
    pk.h[0] = (bf16_t)tv.x; pk.h[1] = (bf16_t)tv.y;
    pk.h[2] = (bf16_t)tv.z; pk.h[3] = (bf16_t)tv.w;
    *(uint2*)(vn + ((size_t)(kvh * CTX_ + s)) * HD_ + d0) = pk.u;
  }
}

// ---------------- Flash attention ------------------------------------------
// grid (S_/16, KVH_), block 128: 4 waves = 4 query heads sharing one KV group,
// same 16 q-rows; K/V tiles async-staged into LDS, double-buffered.
constexpr int LDK_S = 32 + 8;    // 40
constexpr int LDV_S = 128 + 8;   // 136
constexpr int LDP_S = 32 + 8;    // 40

__global__ __launch_bounds__(128)
void attn_kernel(const bf16_t* __restrict__ qn,  // [H][S][HD]
                 const bf16_t* __restrict__ kt,  // [KVH][HD][CTX]
                 const bf16_t* __restrict__ vn,  // [KVH][CTX][HD]
                 bf16_t* __restrict__ yb) {      // [S][H*HD]
  __shared__ __attribute__((aligned(16))) bf16_t Kb[2][HD_ * LDK_S];
  __shared__ __attribute__((aligned(16))) bf16_t Vb[2][32 * LDV_S];
  __shared__ __attribute__((aligned(16))) bf16_t pbuf[4][16 * LDP_S];
  int wave = threadIdx.x >> 5;
  int lane = threadIdx.x & 31;
  int m0   = blockIdx.x * 16;
  int kvh  = blockIdx.y;
  int h    = kvh * 4 + wave;                     // N_REP = 4

  const bf16_t* qbase = qn + (size_t)h * S_ * HD_;
  const bf16_t* kbase = kt + (size_t)kvh * HD_ * CTX_;
  const bf16_t* vbase = vn + (size_t)kvh * CTX_ * HD_;

  auto copy_kv = [&](int buf, int t0) {
    int t = threadIdx.x;                         // 0..127
#pragma unroll
    for (int i = 0; i < 4; ++i) {
      int c = t + i * 128;                       // 0..511
      int rk = c >> 2, sk = (c & 3) << 3;        // K: 128 rows x 32 elems
      async_copy16(kbase + (size_t)rk * CTX_ + t0 + sk, &Kb[buf][rk * LDK_S + sk]);
      int rv = c >> 4, sv = (c & 15) << 3;       // V: 32 rows x 128 elems
      async_copy16(vbase + (size_t)(t0 + rv) * HD_ + sv, &Vb[buf][rv * LDV_S + sv]);
    }
  };

  v16bf qa[4];
#pragma unroll
  for (int kk = 0; kk < 4; ++kk) qa[kk] = fragA(qbase, HD_, m0, kk * 32);

  v8f o[8];
#pragma unroll
  for (int j = 0; j < 8; ++j) o[j] = vzero8();
  float mrow[8], lrow[8];
#pragma unroll
  for (int r = 0; r < 8; ++r) { mrow[r] = -1e30f; lrow[r] = 0.f; }

  int rowadd = (lane >> 4) << 3;
  int coll   = lane & 15;
  const float scl = 0.08838834764831845f;        // 1/sqrt(128)

  copy_kv(0, 0);
  async_wait_all();
  __syncthreads();

  int buf = 0;
  for (int t0 = 0; t0 <= m0; t0 += 32) {         // causal: keys <= m0+15
    if (t0 + 32 <= m0) copy_kv(buf ^ 1, t0 + 32);

    v8f sc0 = vzero8(), sc1 = vzero8();
#pragma unroll
    for (int kk = 0; kk < 4; ++kk) {
      v16bf b0 = fragB(&Kb[buf][0], LDK_S, kk * 32, 0);
      v16bf b1 = fragB(&Kb[buf][0], LDK_S, kk * 32, 16);
      sc0 = wmma_bf16(qa[kk], b0, sc0);
      sc1 = wmma_bf16(qa[kk], b1, sc1);
    }
    int col0 = t0 + coll;
#pragma unroll
    for (int r = 0; r < 8; ++r) {
      int row = m0 + rowadd + r;
      float x0 = (col0 <= row)      ? sc0[r] * scl : -1e30f;
      float x1 = (col0 + 16 <= row) ? sc1[r] * scl : -1e30f;
      float mx = fmaxf(x0, x1);
#pragma unroll
      for (int d2 = 1; d2 < 16; d2 <<= 1) mx = fmaxf(mx, __shfl_xor(mx, d2, 32));
      float mnew = fmaxf(mrow[r], mx);
      float corr = __expf(mrow[r] - mnew);
      float p0 = __expf(x0 - mnew);
      float p1 = __expf(x1 - mnew);
      float ps = p0 + p1;
#pragma unroll
      for (int d2 = 1; d2 < 16; d2 <<= 1) ps += __shfl_xor(ps, d2, 32);
      lrow[r] = lrow[r] * corr + ps;
      mrow[r] = mnew;
#pragma unroll
      for (int j = 0; j < 8; ++j) o[j][r] *= corr;
      pbuf[wave][(rowadd + r) * LDP_S + coll]      = (bf16_t)p0;
      pbuf[wave][(rowadd + r) * LDP_S + coll + 16] = (bf16_t)p1;
    }
    // repack P (16x32) from per-wave LDS scratch into A-fragment layout
    v16bf pa = fragA(&pbuf[wave][0], LDP_S, 0, 0);
#pragma unroll
    for (int j = 0; j < 8; ++j) {
      v16bf vb = fragB(&Vb[buf][0], LDV_S, 0, j * 16);
      o[j] = wmma_bf16(pa, vb, o[j]);
    }

    async_wait_all();
    __syncthreads();
    buf ^= 1;
  }

#pragma unroll
  for (int r = 0; r < 8; ++r) {
    int row = m0 + rowadd + r;
    float inv = 1.0f / lrow[r];
#pragma unroll
    for (int j = 0; j < 8; ++j) {
      yb[(size_t)row * D_ + h * HD_ + j * 16 + coll] = (bf16_t)(o[j][r] * inv);
    }
  }
}

// ---------------- launcher -------------------------------------------------
extern "C" void kernel_launch(void* const* d_in, const int* in_sizes, int n_in,
                              void* d_out, int out_size, void* d_ws, size_t ws_size,
                              hipStream_t stream) {
  (void)in_sizes; (void)n_in; (void)out_size; (void)ws_size;
  const float* x    = (const float*)d_in[0];
  const float* fcos = (const float*)d_in[1];
  const float* fsin = (const float*)d_in[2];
  const float* wq   = (const float*)d_in[4];
  const float* wk   = (const float*)d_in[5];
  const float* wv   = (const float*)d_in[6];
  const float* wo   = (const float*)d_in[7];
  const float* qw   = (const float*)d_in[8];
  const float* kw   = (const float*)d_in[9];

  char* p = (char*)d_ws;
  auto alloc = [&](size_t bytes) -> void* {
    void* r = (void*)p;
    p += (bytes + 255) & ~(size_t)255;
    return r;
  };
  bf16_t* xb  = (bf16_t*)alloc((size_t)S_ * D_ * 2);
  bf16_t* wc  = (bf16_t*)alloc((size_t)D_ * NQKV_ * 2);   // [wq|wk|wv] bf16
  bf16_t* woc = (bf16_t*)alloc((size_t)D_ * D_ * 2);
  float*  qkv = (float*) alloc((size_t)S_ * NQKV_ * 4);
  bf16_t* qnb = (bf16_t*)alloc((size_t)H_ * S_ * HD_ * 2);
  bf16_t* ktb = (bf16_t*)alloc((size_t)KVH_ * HD_ * CTX_ * 2);
  bf16_t* vnb = (bf16_t*)alloc((size_t)KVH_ * CTX_ * HD_ * 2);
  bf16_t* ybb = (bf16_t*)alloc((size_t)S_ * D_ * 2);

  auto nblk = [](int n) { return (n + 255) / 256; };

  // 1) casts
  cvt_bf16_kernel<<<nblk(S_ * D_), 256, 0, stream>>>(x,  xb,  S_ * D_, D_, D_, 0);
  cvt_bf16_kernel<<<nblk(D_ * D_), 256, 0, stream>>>(wq, wc,  D_ * D_, D_, NQKV_, 0);
  cvt_bf16_kernel<<<nblk(D_ * KVH_ * HD_), 256, 0, stream>>>(wk, wc, D_ * KVH_ * HD_,
                                                             KVH_ * HD_, NQKV_, D_);
  cvt_bf16_kernel<<<nblk(D_ * KVH_ * HD_), 256, 0, stream>>>(wv, wc, D_ * KVH_ * HD_,
                                                             KVH_ * HD_, NQKV_, D_ + KVH_ * HD_);
  cvt_bf16_kernel<<<nblk(D_ * D_), 256, 0, stream>>>(wo, woc, D_ * D_, D_, D_, 0);

  // 2) fused QKV projection (WMMA, async-LDS pipelined)
  gemm_bf16_kernel<<<dim3(NQKV_ / 128, S_ / 128), 256, 0, stream>>>(
      xb, wc, qkv, S_, NQKV_, D_);

  // 3) RoPE + RMSNorm + k/v cache layouts
  rope_norm_kernel<<<dim3(6, S_), 256, 0, stream>>>(qkv, fcos, fsin, qw, kw,
                                                    qnb, ktb, vnb);

  // 4) flash attention (WMMA, shared KV tiles per GQA group)
  attn_kernel<<<dim3(S_ / 16, KVH_), 128, 0, stream>>>(qnb, ktb, vnb, ybb);

  // 5) output projection (WMMA) -> d_out f32
  gemm_bf16_kernel<<<dim3(D_ / 128, S_ / 128), 256, 0, stream>>>(
      ybb, woc, (float*)d_out, S_, D_, D_);
}
